// MCTransformer_55602646614549
// MI455X (gfx1250) — compile-verified, matrix operand order checked by reading, not run on hardware
//
#include <hip/hip_runtime.h>

typedef __attribute__((ext_vector_type(16))) __bf16 v16bf;
typedef __attribute__((ext_vector_type(8)))  __bf16 v8bf;
typedef __attribute__((ext_vector_type(8)))  float  v8f;
typedef __attribute__((ext_vector_type(4)))  float  v4f;
typedef __attribute__((ext_vector_type(4)))  unsigned int v4u;
typedef __attribute__((ext_vector_type(8)))  int    v8i;
typedef __attribute__((ext_vector_type(4)))  int    v4i;

#define BATCH 8192
#define TT    10
#define DD    1024
#define MROWS (BATCH * TT)          // 81920 rows in the flattened projection GEMM
#define NQKV  (3 * DD)              // 3072 concatenated output columns (Q|K|V)

union bfrag { v16bf v; v8bf h[2]; };

// ---------------------------------------------------------------------------
// Kernel 0: convert the three 1024x1024 fp32 weights to one concatenated bf16
// buffer Wb[3][1024][1024] (row-major W[e][d]); 6 MB -> resident in 192MB L2.
// ---------------------------------------------------------------------------
__global__ void wconv_kernel(const float* __restrict__ wq,
                             const float* __restrict__ wk,
                             const float* __restrict__ wv,
                             __bf16* __restrict__ wb) {
  int i   = blockIdx.x * 256 + threadIdx.x;       // 0 .. 3*2^20-1
  int sel = i >> 20;
  int off = i & ((1 << 20) - 1);
  const float* src = (sel == 0) ? wq : (sel == 1) ? wk : wv;
  wb[i] = (__bf16)src[off];
}

// ---------------------------------------------------------------------------
// Kernel 1: fused (x+pe)->bf16 QKV projection GEMM, bias+ReLU(+1/32) epilogue.
// Block = 256 threads = 8 waves.  Block tile: 32 rows x 256 cols of the
// 81920 x 3072 output.  A tile (32x1024 bf16 = 64KB) staged once in LDS.
// Each wave owns a 32x32 output tile: 4 accumulators, 2 A frags (LDS) +
// 2 B frags (global) -> 4 WMMAs per K step = 1 ds_load_b128 + 1
// global_load_b128 per v_wmma_f32_16x16x32_bf16.  grid=(12 ngroups fastest,
// 2560 mtiles) so blocks sharing an A tile hit L2 on the same slice of x.
// ---------------------------------------------------------------------------
__global__ __launch_bounds__(256)
void qkv_gemm_kernel(const float* __restrict__ x,
                     const float* __restrict__ pe,
                     const __bf16* __restrict__ Wb,
                     const float* __restrict__ bq,
                     const float* __restrict__ bk,
                     const float* __restrict__ bv,
                     __bf16* __restrict__ Q,
                     __bf16* __restrict__ K,
                     __bf16* __restrict__ V) {
  __shared__ __align__(16) __bf16 As[32 * DD];    // 64 KB

  const int ngroup = blockIdx.x;                  // 0..11
  const int mtile  = blockIdx.y;                  // 0..2559
  const int row0   = mtile * 32;                  // first global row of tile
  const int tid    = threadIdx.x;

  // Stage A tile: (x + pe) -> bf16, row-major [32][1024].  float4 loads.
  for (int i = tid * 4; i < 32 * DD; i += 256 * 4) {
    int r = i >> 10, d = i & (DD - 1);
    int grow = row0 + r;
    int t = grow % TT;
    v4f xv = *(const v4f*)&x[(size_t)grow * DD + d];
    v4f pv = *(const v4f*)&pe[t * DD + d];
    As[i + 0] = (__bf16)(xv[0] + pv[0]);
    As[i + 1] = (__bf16)(xv[1] + pv[1]);
    As[i + 2] = (__bf16)(xv[2] + pv[2]);
    As[i + 3] = (__bf16)(xv[3] + pv[3]);
  }
  __syncthreads();

  const int wave  = tid >> 5;
  const int lane  = tid & 31;
  const int col   = ngroup * 256 + wave * 32 + (lane & 15); // B column 0
  const int arow  = lane & 15;                              // A row 0
  const int khalf = (lane >> 4) * 8;                        // 0 or 8

  const __bf16* __restrict__ wcol0 = Wb + (size_t)col * DD;
  const __bf16* __restrict__ wcol1 = wcol0 + (size_t)16 * DD;
  const __bf16* __restrict__ arow0 = &As[arow * DD];
  const __bf16* __restrict__ arow1 = &As[(arow + 16) * DD];

  v8f acc00 = {}, acc01 = {}, acc10 = {}, acc11 = {};
  #pragma unroll 2
  for (int k0 = 0; k0 < DD; k0 += 32) {
    bfrag a0, a1, b0, b1;
    a0.h[0] = *(const v8bf*)&arow0[k0 + khalf];
    a0.h[1] = *(const v8bf*)&arow0[k0 + khalf + 16];
    a1.h[0] = *(const v8bf*)&arow1[k0 + khalf];
    a1.h[1] = *(const v8bf*)&arow1[k0 + khalf + 16];
    b0.h[0] = *(const v8bf*)&wcol0[k0 + khalf];
    b0.h[1] = *(const v8bf*)&wcol0[k0 + khalf + 16];
    b1.h[0] = *(const v8bf*)&wcol1[k0 + khalf];
    b1.h[1] = *(const v8bf*)&wcol1[k0 + khalf + 16];
    acc00 = __builtin_amdgcn_wmma_f32_16x16x32_bf16(false, a0.v, false, b0.v,
                                                    (short)0, acc00, false, false);
    acc01 = __builtin_amdgcn_wmma_f32_16x16x32_bf16(false, a0.v, false, b1.v,
                                                    (short)0, acc01, false, false);
    acc10 = __builtin_amdgcn_wmma_f32_16x16x32_bf16(false, a1.v, false, b0.v,
                                                    (short)0, acc10, false, false);
    acc11 = __builtin_amdgcn_wmma_f32_16x16x32_bf16(false, a1.v, false, b1.v,
                                                    (short)0, acc11, false, false);
  }

  // Epilogue: bias + ReLU (+ 1/sqrt(D) folded into Q), store bf16.
  const int w = col >> 10;                        // 0=Q 1=K 2=V (wave-uniform)
  const int e = col & (DD - 1);
  const float bias0 = (w == 0) ? bq[e]      : (w == 1) ? bk[e]      : bv[e];
  const float bias1 = (w == 0) ? bq[e + 16] : (w == 1) ? bk[e + 16] : bv[e + 16];
  const float scale = (w == 0) ? 0.03125f : 1.0f; // D^-0.5 = 1/32
  __bf16* __restrict__ outp = (w == 0) ? Q : (w == 1) ? K : V;

  const int m0 = (lane < 16) ? 0 : 8;             // C/D layout: VGPR r -> row m0+r
  #pragma unroll
  for (int i = 0; i < 8; ++i) {
    size_t ra = (size_t)(row0 + m0 + i) * DD;
    size_t rb = (size_t)(row0 + 16 + m0 + i) * DD;
    float v00 = fmaxf(acc00[i] + bias0, 0.0f) * scale;
    float v01 = fmaxf(acc01[i] + bias1, 0.0f) * scale;
    float v10 = fmaxf(acc10[i] + bias0, 0.0f) * scale;
    float v11 = fmaxf(acc11[i] + bias1, 0.0f) * scale;
    outp[ra + e]      = (__bf16)v00;
    outp[ra + e + 16] = (__bf16)v01;
    outp[rb + e]      = (__bf16)v10;
    outp[rb + e + 16] = (__bf16)v11;
  }
}

// ---------------------------------------------------------------------------
// Kernel 2: per-batch attention + residual + LayerNorm.  One block per batch.
// Wave 0: scores = Q·K^T via 32 bf16 WMMAs (valid 10x10 inside 16x16 tile).
// Wave 1 concurrently issues a Tensor Data Mover DMA staging the batch's V
// rows (contiguous 10x1024 bf16 = 20KB) into LDS, then s_wait_tensorcnt.
// Then softmax, attn -> d_out tail, attn@V + residual in LDS, LN per row.
// ---------------------------------------------------------------------------
__global__ __launch_bounds__(256)
void attn_ln_kernel(const float* __restrict__ x,
                    const float* __restrict__ pe,
                    const __bf16* __restrict__ Q,
                    const __bf16* __restrict__ K,
                    const __bf16* __restrict__ V,
                    const float* __restrict__ gamma,
                    const float* __restrict__ beta,
                    float* __restrict__ out_x,
                    float* __restrict__ out_attn) {
  __shared__ __align__(16) __bf16 Vs[TT * DD];    // 20 KB (TDM destination)
  __shared__ float attn_s[16][16];                // 1 KB
  __shared__ float ys[TT][DD];                    // 40 KB

  const int b    = blockIdx.x;
  const int tid  = threadIdx.x;
  const int wave = tid >> 5;
  const int lane = tid & 31;
  const size_t rbase = (size_t)b * TT;            // first row of this batch

  if (wave == 0) {
    // Scores tile.  Rows/cols >= 10 clamped to 9 (padding, unused after).
    int idx   = lane & 15;
    int rc    = (idx > 9) ? 9 : idx;
    int khalf = (lane >> 4) * 8;
    const __bf16* __restrict__ qrow = Q + (rbase + rc) * DD;
    const __bf16* __restrict__ krow = K + (rbase + rc) * DD;
    v8f acc = {};
    #pragma unroll 4
    for (int k0 = 0; k0 < DD; k0 += 32) {
      bfrag a, bf;
      a.h[0]  = *(const v8bf*)&qrow[k0 + khalf];
      a.h[1]  = *(const v8bf*)&qrow[k0 + khalf + 16];
      bf.h[0] = *(const v8bf*)&krow[k0 + khalf];
      bf.h[1] = *(const v8bf*)&krow[k0 + khalf + 16];
      acc = __builtin_amdgcn_wmma_f32_16x16x32_bf16(false, a.v, false, bf.v,
                                                    (short)0, acc, false, false);
    }
    int m0 = (lane < 16) ? 0 : 8;
    #pragma unroll
    for (int i = 0; i < 8; ++i) attn_s[m0 + i][idx] = acc[i];
  } else if (wave == 1) {
    // TDM: stage V rows of this batch into LDS (1-D tile, 10240 bf16 elems).
    // D# per CDNA5 ISA ch.8: group0 = {flags, lds_addr, global_addr, type},
    // group1 = dims/strides, groups 2-3 unused (zero).
    unsigned long long ga = (unsigned long long)(const void*)(V + rbase * DD);
    unsigned int lds     = (unsigned int)(unsigned long long)(void*)&Vs[0];
    const unsigned int nelem = TT * DD;           // 10240
    v4u g0 = { 1u,                                // count=1 valid descriptor
               lds,                               // lds_addr (bytes)
               (unsigned int)(ga & 0xffffffffu),  // global_addr[31:0]
               (unsigned int)((ga >> 32) & 0x1ffffffu) | (2u << 30) }; // type=2
    v8i g1 = { (int)(1u << 16),                   // data_size=1 (2B), mask=0
               (int)((nelem & 0xffffu) << 16),    // tensor_dim0[15:0]
               (int)(((nelem >> 16) & 0xffffu) | (1u << 16)), // dim0 hi, dim1=1
               (int)((nelem & 0xffffu) << 16),    // tile_dim0 = 10240
               0,                                 // tile_dim1/2 unused
               (int)nelem,                        // tensor_dim0_stride lo32
               0, 0 };
    v4i g2 = { 0, 0, 0, 0 };
    v4i g3 = { 0, 0, 0, 0 };
#if __has_include(<hip/amd_detail/amd_gfx1250_TDM.h>)
    v8i g4 = { 0, 0, 0, 0, 0, 0, 0, 0 };
    __builtin_amdgcn_tensor_load_to_lds(g0, g1, g2, g3, g4, 0);
#else
    __builtin_amdgcn_tensor_load_to_lds(g0, g1, g2, g3, 0);
#endif
    __builtin_amdgcn_s_wait_tensorcnt(0);
  }
  __syncthreads();

  // Softmax over the 10 valid columns; one thread per row.
  if (tid < TT) {
    int t = tid;
    float m = attn_s[t][0];
    #pragma unroll
    for (int s = 1; s < TT; ++s) m = fmaxf(m, attn_s[t][s]);
    float p[TT], sum = 0.0f;
    #pragma unroll
    for (int s = 0; s < TT; ++s) { p[s] = __expf(attn_s[t][s] - m); sum += p[s]; }
    float inv = 1.0f / sum;
    #pragma unroll
    for (int s = 0; s < TT; ++s) {
      float pv = p[s] * inv;
      attn_s[t][s] = pv;
      out_attn[(size_t)b * TT * TT + t * TT + s] = pv;
    }
  }
  __syncthreads();

  // out = attn @ V ; residual y = (x + pe) + out, staged in LDS for LN.
  for (int i = tid; i < TT * DD; i += 256) {
    int t = i >> 10, d = i & (DD - 1);
    float o = 0.0f;
    #pragma unroll
    for (int s = 0; s < TT; ++s) o += attn_s[t][s] * (float)Vs[s * DD + d];
    float xv = x[(rbase + t) * DD + d] + pe[t * DD + d];
    ys[t][d] = xv + o;
  }
  __syncthreads();

  // LayerNorm per row; one wave per row (waves 0,1 take two rows).
  for (int t = wave; t < TT; t += 8) {
    float s1 = 0.0f, s2 = 0.0f;
    for (int d = lane; d < DD; d += 32) {
      float v = ys[t][d];
      s1 += v; s2 += v * v;
    }
    #pragma unroll
    for (int off = 16; off > 0; off >>= 1) {
      s1 += __shfl_xor(s1, off);
      s2 += __shfl_xor(s2, off);
    }
    float mu   = s1 * (1.0f / DD);
    float var  = s2 * (1.0f / DD) - mu * mu;
    float rstd = rsqrtf(var + 1e-5f);
    for (int d = lane; d < DD; d += 32) {
      out_x[(rbase + t) * DD + d] = (ys[t][d] - mu) * rstd * gamma[d] + beta[d];
    }
  }
}

// ---------------------------------------------------------------------------
extern "C" void kernel_launch(void* const* d_in, const int* in_sizes, int n_in,
                              void* d_out, int out_size, void* d_ws, size_t ws_size,
                              hipStream_t stream) {
  const float* x     = (const float*)d_in[0];
  const float* pe    = (const float*)d_in[1];
  const float* Wq    = (const float*)d_in[2];
  const float* bq    = (const float*)d_in[3];
  const float* Wk    = (const float*)d_in[4];
  const float* bk    = (const float*)d_in[5];
  const float* Wv    = (const float*)d_in[6];
  const float* bv    = (const float*)d_in[7];
  const float* gamma = (const float*)d_in[8];
  const float* beta  = (const float*)d_in[9];

  float* out_x    = (float*)d_out;
  float* out_attn = out_x + (size_t)MROWS * DD;   // attn tail of d_out

  // Workspace layout (bf16 buffers), ~486 MiB total.
  char* ws = (char*)d_ws;
  __bf16* Wb = (__bf16*)ws;                                   //   6 MiB
  __bf16* Q  = (__bf16*)(ws + (size_t)NQKV * DD * 2);          // 160 MiB
  __bf16* K  = Q + (size_t)MROWS * DD;                         // 160 MiB
  __bf16* V  = K + (size_t)MROWS * DD;                         // 160 MiB

  wconv_kernel<<<(3 << 20) / 256, 256, 0, stream>>>(Wq, Wk, Wv, Wb);

  dim3 g1(NQKV / 256, MROWS / 32);                // (12, 2560), ngroup fastest
  qkv_gemm_kernel<<<g1, 256, 0, stream>>>(x, pe, Wb, bq, bk, bv, Q, K, V);

  attn_ln_kernel<<<BATCH, 256, 0, stream>>>(x, pe, Q, K, V, gamma, beta,
                                            out_x, out_attn);
}
